// DecoderOnlyLayer_67302137528535
// MI455X (gfx1250) — matrix operand order repaired and checked by reading, not compiled, not run on hardware
//
#include <hip/hip_runtime.h>

// ---------------------------------------------------------------------------
// DecoderOnlyLayer for MI455X (gfx1250, wave32, WMMA + Tensor Data Mover).
// GEMMs + attention via v_wmma_f32_16x16x32_f16; GEMM tiles staged into LDS
// with tensor_load_to_lds (TENSORcnt double buffering) when available.
// ---------------------------------------------------------------------------

typedef __attribute__((ext_vector_type(16))) _Float16 v16h;
typedef __attribute__((ext_vector_type(8)))  _Float16 v8h;
typedef __attribute__((ext_vector_type(8)))  float    v8f;
typedef __attribute__((ext_vector_type(4)))  float    v4f;
typedef __attribute__((ext_vector_type(4)))  unsigned int v4u;
typedef __attribute__((ext_vector_type(8)))  int      v8i;
typedef __attribute__((ext_vector_type(4)))  int      v4i;

#define DM   1024
#define NH   16
#define HD   64
#define DFF  4096
#define BB   2
#define SS   2048
#define MTOT (BB * SS)   // 4096 rows total

#if defined(__has_builtin)
#if __has_builtin(__builtin_amdgcn_tensor_load_to_lds) && \
    __has_builtin(__builtin_amdgcn_s_wait_tensorcnt)
#define USE_TDM 1
#endif
#endif
#ifndef USE_TDM
#define USE_TDM 0
#endif

static __device__ inline v8f wmma_f16(const v16h& a, const v16h& b, const v8f& c) {
    return __builtin_amdgcn_wmma_f32_16x16x32_f16(
        /*neg_a=*/false, a, /*neg_b=*/false, b,
        /*c_mod=*/(short)0, c, /*reuse_a=*/false, /*reuse_b=*/false);
}

static __device__ inline v16h pack16(const v8h& lo, const v8h& hi) {
    v16h r;
#pragma unroll
    for (int i = 0; i < 8; ++i) { r[i] = lo[i]; r[i + 8] = hi[i]; }
    return r;
}

#if USE_TDM
// Issue a 2D TDM tile load: tile_h rows x tile_w f16 elems, global row
// stride = rowlen elems, into LDS at lds_off with +16B pad per 64B row
// (pad_interval=16 dwords, pad_amount=4 dwords -> 80B LDS row stride).
static __device__ inline void tdm_load_2d(unsigned lds_off, const _Float16* g,
                                          unsigned rowlen, unsigned nrows,
                                          unsigned tile_w, unsigned tile_h) {
    unsigned long long ga = (unsigned long long)(uintptr_t)g;
    v4u g0;
    g0[0] = 1u;                                   // count=1 (valid user D#)
    g0[1] = lds_off;                              // LDS byte address
    g0[2] = (unsigned)ga;                         // global addr lo
    g0[3] = (unsigned)(ga >> 32) | (2u << 30);    // global addr hi | type=2
    v8i g1;
    g1[0] = (int)((1u << 16)                      // data_size = 2 bytes
                | (1u << 20)                      // pad_enable
                | (3u << 22)                      // pad_interval = 16 dwords
                | (3u << 25));                    // pad_amount   = 4 dwords
    g1[1] = (int)((rowlen & 0xFFFFu) << 16);                       // td0.lo
    g1[2] = (int)((rowlen >> 16) | ((nrows & 0xFFFFu) << 16));     // td0.hi|td1.lo
    g1[3] = (int)((nrows >> 16) | (tile_w << 16));                 // td1.hi|tile0
    g1[4] = (int)(tile_h & 0xFFFFu);                               // tile1|tile2=0
    g1[5] = (int)rowlen;                          // tensor_dim0_stride lo
    g1[6] = 0;                                    // stride hi / dim1_stride
    g1[7] = 0;
    v4i z;  z[0] = z[1] = z[2] = z[3] = 0;
#if __clang_major__ >= 23
    v8i z8;
#pragma unroll
    for (int i = 0; i < 8; ++i) z8[i] = 0;
    __builtin_amdgcn_tensor_load_to_lds(g0, g1, z, z, z8, 0);
#else
    __builtin_amdgcn_tensor_load_to_lds(g0, g1, z, z, 0);
#endif
}
#endif

// ---------------------------------------------------------------------------
// f32 -> f16 conversion (4 elems/thread)
// ---------------------------------------------------------------------------
__global__ void __launch_bounds__(256)
k_cvt_f16(const float* __restrict__ in, _Float16* __restrict__ out, int n) {
    int i = (blockIdx.x * 256 + threadIdx.x) * 4;
    if (i + 3 < n) {
        v4f v = *(const v4f*)(in + i);
        out[i + 0] = (_Float16)v[0];
        out[i + 1] = (_Float16)v[1];
        out[i + 2] = (_Float16)v[2];
        out[i + 3] = (_Float16)v[3];
    }
}

// ---------------------------------------------------------------------------
// f32 [K,N] -> f16 transposed [N,K] (32x32 LDS-tiled)
// ---------------------------------------------------------------------------
__global__ void __launch_bounds__(256)
k_cvt_f16_t(const float* __restrict__ in, _Float16* __restrict__ out,
            int K, int N) {
    __shared__ float tile[32][33];
    int bx = blockIdx.x * 32;                 // along N (input cols)
    int by = blockIdx.y * 32;                 // along K (input rows)
    int tx = threadIdx.x & 31, ty = threadIdx.x >> 5;   // 32 x 8
#pragma unroll
    for (int i = 0; i < 32; i += 8)
        tile[ty + i][tx] = in[(size_t)(by + ty + i) * N + bx + tx];
    __syncthreads();
#pragma unroll
    for (int i = 0; i < 32; i += 8)
        out[(size_t)(bx + ty + i) * K + by + tx] = (_Float16)tile[tx][ty + i];
}

// ---------------------------------------------------------------------------
// [B,S,H*64] -> Q/K: [B,H,S,64], V -> V^T: [B,H,64,S]
// ---------------------------------------------------------------------------
__global__ void __launch_bounds__(256)
k_permute_qkv(const _Float16* __restrict__ q, const _Float16* __restrict__ k,
              const _Float16* __restrict__ v,
              _Float16* __restrict__ Qh, _Float16* __restrict__ Kh,
              _Float16* __restrict__ VTh) {
    int idx = blockIdx.x * 256 + threadIdx.x;      // over B*S*DM
    int c  = idx % DM;
    int bs = idx / DM;
    int s  = bs % SS;
    int b  = bs / SS;
    int h  = c / HD, d = c % HD;
    int bh = b * NH + h;
    Qh [((size_t)bh * SS + s) * HD + d] = q[idx];
    Kh [((size_t)bh * SS + s) * HD + d] = k[idx];
    VTh[((size_t)bh * HD + d) * SS + s] = v[idx];
}

// ---------------------------------------------------------------------------
// WMMA GEMM: out = A[M,K](f16) @ BT[N,K](f16)^T + bias (+resid) (ReLU?)
// Block tile 128x128, K-tile 32. 8 waves = 4(M) x 2(N), each wave 32x64.
// Both A and BT tiles are row-major 2D tiles -> staged by TDM
// (tensor_load_to_lds, TENSORcnt) with double buffering; manual fallback.
// LDS row stride 40 halves (80B) via TDM pad fields.
// ---------------------------------------------------------------------------
template <bool RELU, bool RESID, bool OUT16>
__global__ void __launch_bounds__(256)
k_gemm_wmma(const _Float16* __restrict__ A, const _Float16* __restrict__ BT,
            const float* __restrict__ bias, const float* __restrict__ resid,
            _Float16* __restrict__ out16, float* __restrict__ out32,
            int M, int N, int K) {
    __shared__ _Float16 As[2][128 * 40];
    __shared__ _Float16 Bs[2][128 * 40];

    const int tid  = threadIdx.x;
    const int lane = tid & 31;
    const int wave = tid >> 5;
    const int wm   = wave & 3;     // 0..3 -> 32-row slice
    const int wn   = wave >> 2;    // 0..1 -> 64-col slice
    const int m0   = blockIdx.y * 128;
    const int n0   = blockIdx.x * 128;
    const int ln   = lane & 15;
    const int lh   = lane >> 4;
    const int nkt  = K >> 5;

    v8f acc[2][4];
#pragma unroll
    for (int mt = 0; mt < 2; ++mt)
#pragma unroll
        for (int nt = 0; nt < 4; ++nt)
#pragma unroll
            for (int i = 0; i < 8; ++i) acc[mt][nt][i] = 0.0f;

#if USE_TDM
    // ---- TDM double-buffered staging ----
    if (wave == 0) {
        tdm_load_2d((unsigned)(uintptr_t)&As[0][0], A  + (size_t)m0 * K,
                    (unsigned)K, (unsigned)M, 32u, 128u);
        tdm_load_2d((unsigned)(uintptr_t)&Bs[0][0], BT + (size_t)n0 * K,
                    (unsigned)K, (unsigned)N, 32u, 128u);
        __builtin_amdgcn_s_wait_tensorcnt(0);
    }
    __syncthreads();

    for (int kt = 0; kt < nkt; ++kt) {
        const int cur = kt & 1;
        if (wave == 0 && kt + 1 < nkt) {
            const int k0n = (kt + 1) << 5;
            tdm_load_2d((unsigned)(uintptr_t)&As[cur ^ 1][0],
                        A  + (size_t)m0 * K + k0n, (unsigned)K, (unsigned)M,
                        32u, 128u);
            tdm_load_2d((unsigned)(uintptr_t)&Bs[cur ^ 1][0],
                        BT + (size_t)n0 * K + k0n, (unsigned)K, (unsigned)N,
                        32u, 128u);
        }
        const _Float16* Asb = &As[cur][0];
        const _Float16* Bsb = &Bs[cur][0];
        v16h af[2], bf[4];
#pragma unroll
        for (int mt = 0; mt < 2; ++mt) {
            int row = wm * 32 + mt * 16 + ln;
            v8h lo = *(const v8h*)(Asb + row * 40 + 8 * lh);
            v8h hi = *(const v8h*)(Asb + row * 40 + 16 + 8 * lh);
            af[mt] = pack16(lo, hi);
        }
#pragma unroll
        for (int nt = 0; nt < 4; ++nt) {
            int col = wn * 64 + nt * 16 + ln;
            v8h lo = *(const v8h*)(Bsb + col * 40 + 8 * lh);
            v8h hi = *(const v8h*)(Bsb + col * 40 + 16 + 8 * lh);
            bf[nt] = pack16(lo, hi);
        }
#pragma unroll
        for (int mt = 0; mt < 2; ++mt)
#pragma unroll
            for (int nt = 0; nt < 4; ++nt)
                acc[mt][nt] = wmma_f16(af[mt], bf[nt], acc[mt][nt]);
        if (kt + 1 < nkt) {
            if (wave == 0) __builtin_amdgcn_s_wait_tensorcnt(0);
            __syncthreads();
        }
    }
#else
    // ---- manual staging fallback (both tiles row-major, identical pattern) --
    const int arow = tid >> 1;
    const int acol = (tid & 1) * 16;
    for (int kt = 0; kt < nkt; ++kt) {
        const int k0 = kt << 5;
        {
            const _Float16* gA = A + (size_t)(m0 + arow) * K + k0 + acol;
            v8h a0 = *(const v8h*)gA;
            v8h a1 = *(const v8h*)(gA + 8);
            *(v8h*)(&As[0][0] + arow * 40 + acol)     = a0;
            *(v8h*)(&As[0][0] + arow * 40 + acol + 8) = a1;
            const _Float16* gB = BT + (size_t)(n0 + arow) * K + k0 + acol;
            v8h b0 = *(const v8h*)gB;
            v8h b1 = *(const v8h*)(gB + 8);
            *(v8h*)(&Bs[0][0] + arow * 40 + acol)     = b0;
            *(v8h*)(&Bs[0][0] + arow * 40 + acol + 8) = b1;
            if (k0 + 32 < K) __builtin_prefetch(gA + 32, 0, 0);
        }
        __syncthreads();
        v16h af[2], bf[4];
#pragma unroll
        for (int mt = 0; mt < 2; ++mt) {
            int row = wm * 32 + mt * 16 + ln;
            v8h lo = *(const v8h*)(&As[0][0] + row * 40 + 8 * lh);
            v8h hi = *(const v8h*)(&As[0][0] + row * 40 + 16 + 8 * lh);
            af[mt] = pack16(lo, hi);
        }
#pragma unroll
        for (int nt = 0; nt < 4; ++nt) {
            int col = wn * 64 + nt * 16 + ln;
            v8h lo = *(const v8h*)(&Bs[0][0] + col * 40 + 8 * lh);
            v8h hi = *(const v8h*)(&Bs[0][0] + col * 40 + 16 + 8 * lh);
            bf[nt] = pack16(lo, hi);
        }
#pragma unroll
        for (int mt = 0; mt < 2; ++mt)
#pragma unroll
            for (int nt = 0; nt < 4; ++nt)
                acc[mt][nt] = wmma_f16(af[mt], bf[nt], acc[mt][nt]);
        __syncthreads();
    }
#endif

    // ---- epilogue ----
#pragma unroll
    for (int mt = 0; mt < 2; ++mt)
#pragma unroll
        for (int nt = 0; nt < 4; ++nt) {
            int col = n0 + wn * 64 + nt * 16 + ln;
            float bv = bias[col];
#pragma unroll
            for (int r = 0; r < 8; ++r) {
                int rowg = m0 + wm * 32 + mt * 16 + r + 8 * lh;
                float v = acc[mt][nt][r] + bv;
                if (RESID) v += resid[(size_t)rowg * N + col];
                if (RELU)  v = v > 0.0f ? v : 0.0f;
                if (OUT16) out16[(size_t)rowg * N + col] = (_Float16)v;
                else       out32[(size_t)rowg * N + col] = v;
            }
        }
}

// ---------------------------------------------------------------------------
// Flash attention, causal. One wave = 16 query rows; 8 waves/block.
// grid = (S/128, NH, B). Q/K in [B,H,S,64], V^T in [B,H,64,S], ctx -> [B,S,DM].
// ---------------------------------------------------------------------------
__global__ void __launch_bounds__(256)
k_flash_attn(const _Float16* __restrict__ Qh, const _Float16* __restrict__ Kh,
             const _Float16* __restrict__ VTh, _Float16* __restrict__ ctx) {
    __shared__ _Float16 Pst[8][16 * 32];   // per-wave P staging tile

    const int tid  = threadIdx.x;
    const int lane = tid & 31;
    const int wave = tid >> 5;
    const int ln   = lane & 15;
    const int lh   = lane >> 4;
    const int b  = blockIdx.z, h = blockIdx.y;
    const int qbase = blockIdx.x * 128 + wave * 16;
    const int bh = b * NH + h;

    const _Float16* Qp = Qh  + (size_t)bh * SS * HD;
    const _Float16* Kp = Kh  + (size_t)bh * SS * HD;
    const _Float16* Vp = VTh + (size_t)bh * HD * SS;
    _Float16* Ps = &Pst[wave][0];

    v16h aq[2];
#pragma unroll
    for (int kd = 0; kd < 2; ++kd) {
        const _Float16* qp = Qp + (size_t)(qbase + ln) * HD + kd * 32 + 8 * lh;
        aq[kd] = pack16(*(const v8h*)qp, *(const v8h*)(qp + 16));
    }

    float mi[8], li[8];
    v8f   ov[4];
#pragma unroll
    for (int r = 0; r < 8; ++r) { mi[r] = -1e30f; li[r] = 0.0f; }
#pragma unroll
    for (int dg = 0; dg < 4; ++dg)
#pragma unroll
        for (int i = 0; i < 8; ++i) ov[dg][i] = 0.0f;

    const float scale = 0.125f;                 // 1/sqrt(64)
    const int nkb = (qbase + 16 + 31) >> 5;     // causal: only needed key blocks

    for (int kb = 0; kb < nkb; ++kb) {
        const int keyb = kb * 32;

        v8f sc[2];
#pragma unroll
        for (int kt = 0; kt < 2; ++kt) {
#pragma unroll
            for (int i = 0; i < 8; ++i) sc[kt][i] = 0.0f;
            const _Float16* kp = Kp + (size_t)(keyb + kt * 16 + ln) * HD;
#pragma unroll
            for (int kd = 0; kd < 2; ++kd) {
                const _Float16* kpp = kp + kd * 32 + 8 * lh;
                v16h bk = pack16(*(const v8h*)kpp, *(const v8h*)(kpp + 16));
                sc[kt] = wmma_f16(aq[kd], bk, sc[kt]);
            }
        }

        float pv[2][8], rm[8], alpha[8], rsum[8];
#pragma unroll
        for (int kt = 0; kt < 2; ++kt) {
            int key = keyb + kt * 16 + ln;
#pragma unroll
            for (int r = 0; r < 8; ++r) {
                int qrow = qbase + r + 8 * lh;
                float v = sc[kt][r] * scale;
                pv[kt][r] = (key > qrow) ? -1e30f : v;
            }
        }
#pragma unroll
        for (int r = 0; r < 8; ++r) rm[r] = fmaxf(pv[0][r], pv[1][r]);
        for (int m = 1; m < 16; m <<= 1)
#pragma unroll
            for (int r = 0; r < 8; ++r)
                rm[r] = fmaxf(rm[r], __shfl_xor(rm[r], m, 32));
#pragma unroll
        for (int r = 0; r < 8; ++r) {
            float mnew = fmaxf(mi[r], rm[r]);
            alpha[r] = __expf(mi[r] - mnew);
            mi[r] = mnew;
            pv[0][r] = __expf(pv[0][r] - mnew);
            pv[1][r] = __expf(pv[1][r] - mnew);
            rsum[r] = pv[0][r] + pv[1][r];
        }
        for (int m = 1; m < 16; m <<= 1)
#pragma unroll
            for (int r = 0; r < 8; ++r)
                rsum[r] += __shfl_xor(rsum[r], m, 32);
#pragma unroll
        for (int r = 0; r < 8; ++r) li[r] = li[r] * alpha[r] + rsum[r];
#pragma unroll
        for (int dg = 0; dg < 4; ++dg)
#pragma unroll
            for (int r = 0; r < 8; ++r) ov[dg][r] *= alpha[r];

        // P (C-layout) -> wave-private LDS tile -> A-fragment layout
#pragma unroll
        for (int kt = 0; kt < 2; ++kt)
#pragma unroll
            for (int r = 0; r < 8; ++r)
                Ps[(r + 8 * lh) * 32 + kt * 16 + ln] = (_Float16)pv[kt][r];
        const _Float16* pr = Ps + ln * 32 + 8 * lh;
        v16h ap = pack16(*(const v8h*)pr, *(const v8h*)(pr + 16));

#pragma unroll
        for (int dg = 0; dg < 4; ++dg) {
            const _Float16* vp = Vp + (size_t)(dg * 16 + ln) * SS + keyb + 8 * lh;
            v16h bv = pack16(*(const v8h*)vp, *(const v8h*)(vp + 16));
            ov[dg] = wmma_f16(ap, bv, ov[dg]);
        }
    }

#pragma unroll
    for (int dg = 0; dg < 4; ++dg)
#pragma unroll
        for (int r = 0; r < 8; ++r) {
            int row = qbase + r + 8 * lh;
            float cv = ov[dg][r] / li[r];
            ctx[((size_t)b * SS + row) * DM + h * HD + dg * 16 + ln] = (_Float16)cv;
        }
}

// ---------------------------------------------------------------------------
// In-place LayerNorm over rows of 1024, one block/row. Optional f16 copy.
// ---------------------------------------------------------------------------
__global__ void __launch_bounds__(256)
k_layernorm(float* __restrict__ x, const float* __restrict__ gamma,
            const float* __restrict__ beta, _Float16* __restrict__ out16) {
    __shared__ float red[8];
    const int row  = blockIdx.x;
    const int tid  = threadIdx.x;
    const int lane = tid & 31;
    const int wave = tid >> 5;
    float* xr = x + (size_t)row * DM;

    v4f v = *(const v4f*)(xr + tid * 4);
    float s = v[0] + v[1] + v[2] + v[3];
    for (int m = 1; m < 32; m <<= 1) s += __shfl_xor(s, m, 32);
    if (lane == 0) red[wave] = s;
    __syncthreads();
    float tot = 0.0f;
#pragma unroll
    for (int i = 0; i < 8; ++i) tot += red[i];
    float mean = tot * (1.0f / DM);
    __syncthreads();

    float d0 = v[0] - mean, d1 = v[1] - mean, d2 = v[2] - mean, d3 = v[3] - mean;
    float s2 = d0 * d0 + d1 * d1 + d2 * d2 + d3 * d3;
    for (int m = 1; m < 32; m <<= 1) s2 += __shfl_xor(s2, m, 32);
    if (lane == 0) red[wave] = s2;
    __syncthreads();
    float tot2 = 0.0f;
#pragma unroll
    for (int i = 0; i < 8; ++i) tot2 += red[i];
    float rstd = rsqrtf(tot2 * (1.0f / DM) + 1e-6f);

#pragma unroll
    for (int j = 0; j < 4; ++j) {
        int c = tid * 4 + j;
        float o = (v[j] - mean) * rstd * gamma[c] + beta[c];
        xr[c] = o;
        if (out16) out16[(size_t)row * DM + c] = (_Float16)o;
    }
}

// ---------------------------------------------------------------------------
// Host-side orchestration
// ---------------------------------------------------------------------------
extern "C" void kernel_launch(void* const* d_in, const int* in_sizes, int n_in,
                              void* d_out, int out_size, void* d_ws, size_t ws_size,
                              hipStream_t stream) {
    (void)in_sizes; (void)n_in; (void)out_size; (void)ws_size;

    const float* x      = (const float*)d_in[0];
    // d_in[1] = mask (unused: causal mask applied analytically)
    const float* Wq = (const float*)d_in[2],  *bq = (const float*)d_in[3];
    const float* Wk = (const float*)d_in[4],  *bk = (const float*)d_in[5];
    const float* Wv = (const float*)d_in[6],  *bv = (const float*)d_in[7];
    const float* Wo = (const float*)d_in[8],  *bo = (const float*)d_in[9];
    const float* W1 = (const float*)d_in[10], *b1 = (const float*)d_in[11];
    const float* W2 = (const float*)d_in[12], *b2 = (const float*)d_in[13];
    const float* gamma1 = (const float*)d_in[14], *beta1 = (const float*)d_in[15];
    const float* gamma2 = (const float*)d_in[16], *beta2 = (const float*)d_in[17];
    float* out = (float*)d_out;

    // ---- workspace layout ----
    char* p = (char*)d_ws;
    auto alloc = [&](size_t bytes) -> void* {
        void* r = (void*)p;
        p += (bytes + 255) & ~(size_t)255;
        return r;
    };
    const size_t MD  = (size_t)MTOT * DM;    // 4M elems
    const size_t DD  = (size_t)DM * DM;      // 1M
    const size_t DF  = (size_t)DM * DFF;     // 4M
    _Float16* xh   = (_Float16*)alloc(MD * 2);
    _Float16* WqT  = (_Float16*)alloc(DD * 2);   // all weights f16, transposed [N,K]
    _Float16* WkT  = (_Float16*)alloc(DD * 2);
    _Float16* WvT  = (_Float16*)alloc(DD * 2);
    _Float16* WoT  = (_Float16*)alloc(DD * 2);
    _Float16* W1T  = (_Float16*)alloc(DF * 2);
    _Float16* W2T  = (_Float16*)alloc(DF * 2);
    _Float16* ctxh = (_Float16*)alloc(MD * 2);
    _Float16* hh   = (_Float16*)alloc(MD * 2);
    float*    hbuf = (float*)   alloc(MD * 4);
    _Float16* qtmp = (_Float16*)alloc(MD * 2);
    _Float16* ktmp = (_Float16*)alloc(MD * 2);
    _Float16* vtmp = (_Float16*)alloc(MD * 2);
    _Float16* Qh   = (_Float16*)alloc(MD * 2);
    _Float16* Kh   = (_Float16*)alloc(MD * 2);
    _Float16* VTh  = (_Float16*)alloc(MD * 2);
    // ff1 (4096x4096 f16, 32MB) aliases qtmp..Qh (dead after attention)
    _Float16* ff1h = qtmp;

    // ---- 1. convert x; convert+transpose weights ----
    k_cvt_f16<<<MD / 1024, 256, 0, stream>>>(x, xh, (int)MD);
    dim3 gT1(DM / 32, DM / 32);
    k_cvt_f16_t<<<gT1, 256, 0, stream>>>(Wq, WqT, DM, DM);
    k_cvt_f16_t<<<gT1, 256, 0, stream>>>(Wk, WkT, DM, DM);
    k_cvt_f16_t<<<gT1, 256, 0, stream>>>(Wv, WvT, DM, DM);
    k_cvt_f16_t<<<gT1, 256, 0, stream>>>(Wo, WoT, DM, DM);
    dim3 gT2(DFF / 32, DM / 32);
    k_cvt_f16_t<<<gT2, 256, 0, stream>>>(W1, W1T, DM, DFF);   // [DM,DFF]->[DFF,DM]
    dim3 gT3(DM / 32, DFF / 32);
    k_cvt_f16_t<<<gT3, 256, 0, stream>>>(W2, W2T, DFF, DM);   // [DFF,DM]->[DM,DFF]

    // ---- 2. QKV projections ----
    dim3 gP(DM / 128, MTOT / 128);   // (8, 32)
    k_gemm_wmma<false, false, true><<<gP, 256, 0, stream>>>(
        xh, WqT, bq, nullptr, qtmp, nullptr, MTOT, DM, DM);
    k_gemm_wmma<false, false, true><<<gP, 256, 0, stream>>>(
        xh, WkT, bk, nullptr, ktmp, nullptr, MTOT, DM, DM);
    k_gemm_wmma<false, false, true><<<gP, 256, 0, stream>>>(
        xh, WvT, bv, nullptr, vtmp, nullptr, MTOT, DM, DM);

    // ---- 3. head-major permute (V pre-transposed) ----
    k_permute_qkv<<<MD / 256, 256, 0, stream>>>(qtmp, ktmp, vtmp, Qh, Kh, VTh);

    // ---- 4. causal flash attention ----
    dim3 gA(SS / 128, NH, BB);       // (16, 16, 2)
    k_flash_attn<<<gA, 256, 0, stream>>>(Qh, Kh, VTh, ctxh);

    // ---- 5. output projection + residual(x) -> hbuf (f32) ----
    k_gemm_wmma<false, true, false><<<gP, 256, 0, stream>>>(
        ctxh, WoT, bo, x, nullptr, hbuf, MTOT, DM, DM);

    // ---- 6. LayerNorm1 in-place, emit f16 copy ----
    k_layernorm<<<MTOT, 256, 0, stream>>>(hbuf, gamma1, beta1, hh);

    // ---- 7. FFN up + ReLU ----
    dim3 gF(DFF / 128, MTOT / 128);  // (32, 32)
    k_gemm_wmma<true, false, true><<<gF, 256, 0, stream>>>(
        hh, W1T, b1, nullptr, ff1h, nullptr, MTOT, DFF, DM);

    // ---- 8. FFN down + residual(h) -> d_out (f32) ----
    k_gemm_wmma<false, true, false><<<gP, 256, 0, stream>>>(
        ff1h, W2T, b2, hbuf, nullptr, out, MTOT, DM, DFF);

    // ---- 9. LayerNorm2 in-place on d_out ----
    k_layernorm<<<MTOT, 256, 0, stream>>>(out, gamma2, beta2, nullptr);
}
